// BSplineLayer_80315888435614
// MI455X (gfx1250) — matrix operand order repaired and checked by reading, not compiled, not run on hardware
//
#include <hip/hip_runtime.h>
#include <hip/hip_bf16.h>

// Problem dims (fixed by reference)
#define BATCH   8192
#define IN_DIM  512
#define OUT_DIM 512
#define NB      13
#define KTOT    8192          // 512 features * 16 padded K-slots (13 bases + silu + 2 zero)
#define NIT     (IN_DIM / 2)  // K-loop iterations (32 K = 2 features per iter)

#define BM  128
#define BN  128
#define LDA 40                // LDS row stride in ushorts: 32 data + 8 pad

// CDNA5 async Global->LDS path (ASYNCcnt), guarded so we fall back cleanly.
#if defined(__HIP_DEVICE_COMPILE__) && defined(__has_builtin)
#  if __has_builtin(__builtin_amdgcn_global_load_async_to_lds_b128)
#    define HAVE_ASYNC_LDS 1
#  else
#    define HAVE_ASYNC_LDS 0
#  endif
#else
#  define HAVE_ASYNC_LDS 0
#endif

typedef __bf16 bf16_t;
typedef __bf16 v16bf __attribute__((ext_vector_type(16)));
typedef float  v8f   __attribute__((ext_vector_type(8)));

struct Q2 { uint4 a, b; };    // 32 bytes == one v16bf fragment

static __device__ __forceinline__ unsigned short f2bf(float f) {
  return __builtin_bit_cast(unsigned short, (bf16_t)f);
}

#if HAVE_ASYNC_LDS
// Builtin parameter types per hipcc diagnostic: int __vector(4) pointers,
// arg0 in global (AS1), arg1 in LDS (AS3), then two immediates (offset, cpol).
typedef int v4i __attribute__((vector_size(16)));
typedef __attribute__((address_space(1))) v4i* gq_p;
typedef __attribute__((address_space(3))) v4i* lq_p;

static __device__ __forceinline__ void wait_async0() {
#  if __has_builtin(__builtin_amdgcn_s_wait_asynccnt)
  __builtin_amdgcn_s_wait_asynccnt(0);
#  else
  asm volatile("s_wait_asynccnt 0" ::: "memory");
#  endif
}
#endif

// ---------------- pass 0a: pack weights into bf16 [OUT][KTOT], K = i*16 + c ----------------
__global__ void pack_weights(const float* __restrict__ coeffs,   // [OUT, IN, 13]
                             const float* __restrict__ bw,       // [OUT, IN]
                             unsigned short* __restrict__ Wt) {  // [OUT, KTOT] bf16
  int idx = blockIdx.x * blockDim.x + threadIdx.x;               // OUT*KTOT threads
  int k = idx & (KTOT - 1);
  int o = idx >> 13;
  int i = k >> 4;
  int c = k & 15;
  float v = 0.0f;
  if (c < NB)       v = coeffs[(o * IN_DIM + i) * NB + c];
  else if (c == 13) v = bw[o * IN_DIM + i];
  Wt[idx] = f2bf(v);
}

// ---------------- pass 0b: transpose x -> xT [IN][BATCH] f32 ----------------
__global__ void transpose_x(const float* __restrict__ x, float* __restrict__ xT) {
  int idx = blockIdx.x * blockDim.x + threadIdx.x;               // BATCH*IN threads
  int b = idx & (BATCH - 1);
  int i = idx >> 13;
  xT[idx] = x[b * IN_DIM + i];                                   // xT[i*BATCH + b]
}

// ---- stage A: one cubic-B-spline basis evaluation per thread -> 16 bf16 LDS slots ----
static __device__ __forceinline__ void stage_a(int it, unsigned short* __restrict__ Abuf,
                                               const float* __restrict__ xT,
                                               int mBase, int sr, int sh) {
  // Uniform-grid de Boor constants; reference's 1e-8 denominator eps folded in.
  const float A1c = 0.4f / (0.4f + 1e-8f);
  const float A2c = 0.4f / (0.8f + 1e-8f);
  const float A3c = 0.4f / (1.2f + 1e-8f);
  const float One1 = 1.0f * A1c;
  const float One2 = 2.0f * A2c;
  const float One3 = 3.0f * A3c;

  const int feat = 2 * it + sh;
  const float* px = &xT[feat * BATCH + mBase + sr];
  const float xv = *px;
  if (feat + 2 < IN_DIM)
    __builtin_prefetch(px + 2 * BATCH, 0, 3);     // next column pair, keep near WGP

  const float xc = fminf(fmaxf(xv, -1.0f + 1e-4f), 1.0f - 1e-4f);
  const float u  = fmaf(xc, 2.5f, 8.0f);          // knot coordinate, in (5.5, 10.5)
  const float fl = floorf(u);
  const int   ki = (int)fl;                       // support window = bases ki-3 .. ki
  const float fr = u - fl;
  // cubic de Boor restricted to the 4 nonzero bases
  const float t1  = A1c * fr;
  const float c3a = t1;
  const float c2a = One1 - t1;
  const float p2 = A2c * fr, q2 = p2 + A2c;
  const float n3 = p2 * c3a;
  const float n2 = q2 * c2a + (One2 - p2) * c3a;
  const float n1 = (One2 - q2) * c2a;
  const float p3 = A3c * fr, q3 = p3 + A3c, r3 = q3 + A3c;
  const float b3 = p3 * n3;
  const float b2 = q3 * n2 + (One3 - p3) * n3;
  const float b1 = r3 * n1 + (One3 - q3) * n2;
  const float b0 = (One3 - r3) * n1;
  // silu via fast reciprocal (v_rcp_f32) instead of exact division
  const float sil = xv * __builtin_amdgcn_rcpf(1.0f + __expf(-xv));

  unsigned short* rowp = Abuf + sr * LDA + sh * 16;
  *(uint4*)(rowp)     = make_uint4(0u, 0u, 0u, 0u);
  *(uint4*)(rowp + 8) = make_uint4(0u, 0u, 0u, 0u);
  const int c0 = ki - 3;                          // in [2,7]; window fits slots 2..10
  rowp[c0 + 0] = f2bf(b0);
  rowp[c0 + 1] = f2bf(b1);
  rowp[c0 + 2] = f2bf(b2);
  rowp[c0 + 3] = f2bf(b3);
  *(unsigned int*)(rowp + 12) = ((unsigned int)f2bf(sil)) << 16;  // slot12=0, slot13=silu
}

// ---- stage B: [BN x 32] bf16 weight tile, 32B per thread ----
static __device__ __forceinline__ void stage_b(int it, unsigned short* __restrict__ Bbuf,
                                               const unsigned short* __restrict__ Wt,
                                               int nBase, int sr, int sh) {
  const unsigned short* wp = Wt + (size_t)(nBase + sr) * KTOT + it * 32 + sh * 16;
  unsigned short* bp = Bbuf + sr * LDA + sh * 16;
#if HAVE_ASYNC_LDS
  __builtin_amdgcn_global_load_async_to_lds_b128((gq_p)wp,       (lq_p)bp,       0, 0);
  __builtin_amdgcn_global_load_async_to_lds_b128((gq_p)(wp + 8), (lq_p)(bp + 8), 0, 0);
#else
  uint4 w0 = *(const uint4*)(wp);
  uint4 w1 = *(const uint4*)(wp + 8);
  *(uint4*)(bp)     = w0;
  *(uint4*)(bp + 8) = w1;
#endif
}

// ---------------- pass 1: fused basis + GEMM + silu + residual ----------------
__global__ __launch_bounds__(256)
void kan_fused_gemm(const float* __restrict__ x,                 // [B, IN] (residual)
                    const float* __restrict__ xT,                // [IN, B]
                    const unsigned short* __restrict__ Wt,       // [OUT, KTOT] bf16
                    float* __restrict__ out) {                   // [B, OUT]
  __shared__ __align__(16) unsigned short As[2][BM * LDA];
  __shared__ __align__(16) unsigned short Bs[2][BN * LDA];

  const int tid  = threadIdx.x;
  const int lane = tid & 31;
  const int w    = tid >> 5;
  const int wm   = w & 3;          // 4 M-waves
  const int wn   = w >> 2;         // 2 N-waves
  const int mBase = blockIdx.x * BM;
  const int nBase = blockIdx.y * BN;

  const int sr = tid & 127;        // staging row/col in tile
  const int sh = tid >> 7;         // staging K-half (feature 0/1 of the pair)

  v8f zero;
  #pragma unroll
  for (int e = 0; e < 8; ++e) zero[e] = 0.0f;
  v8f acc[2][4];
  #pragma unroll
  for (int a = 0; a < 2; ++a)
    #pragma unroll
    for (int b = 0; b < 4; ++b) acc[a][b] = zero;

  // ---- prologue: stage iteration 0 into buffer 0
  stage_b(0, Bs[0], Wt, nBase, sr, sh);
  stage_a(0, As[0], xT, mBase, sr, sh);
#if HAVE_ASYNC_LDS
  wait_async0();
#endif
  __syncthreads();

  for (int it = 0; it < NIT; ++it) {
    const int cur = it & 1;
    // ---- stage next tile into the other buffer (overlaps with WMMA below)
    if (it + 1 < NIT) {
      stage_b(it + 1, Bs[cur ^ 1], Wt, nBase, sr, sh);
      stage_a(it + 1, As[cur ^ 1], xT, mBase, sr, sh);
    }

    // ---- 2x4 WMMA tiles per wave (wave tile 32x64) on the current buffer
    const int hs = lane >> 4;                     // 0: lanes 0-15, 1: lanes 16-31
    v16bf afrag[2];
    #pragma unroll
    for (int mf = 0; mf < 2; ++mf) {
      // A layout: lane m, K 0-15 (lanes<16) / K 16-31 (lanes>=16), contiguous
      const unsigned short* ap = &As[cur][(wm * 32 + mf * 16 + (lane & 15)) * LDA + hs * 16];
      Q2 q; q.a = *(const uint4*)(ap); q.b = *(const uint4*)(ap + 8);
      afrag[mf] = __builtin_bit_cast(v16bf, q);
    }
    #pragma unroll
    for (int nf = 0; nf < 4; ++nf) {
      // B layout: lane n, K 0-15 (lanes<16) / K 16-31 (lanes>=16), contiguous
      const unsigned short* bp = &Bs[cur][(wn * 64 + nf * 16 + (lane & 15)) * LDA + hs * 16];
      Q2 q; q.a = *(const uint4*)(bp); q.b = *(const uint4*)(bp + 8);
      const v16bf bfrag = __builtin_bit_cast(v16bf, q);
      #pragma unroll
      for (int mf = 0; mf < 2; ++mf) {
        acc[mf][nf] = __builtin_amdgcn_wmma_f32_16x16x32_bf16(
            false, afrag[mf], false, bfrag, (short)0, acc[mf][nf], false, false);
      }
    }

#if HAVE_ASYNC_LDS
    wait_async0();                 // next tile's async B loads landed in LDS
#endif
    __syncthreads();               // one barrier per iteration (double-buffered)
  }

  // ---- epilogue: D lane map (VGPR v: M=v / M=8+v; lanes<16 N=lane) + residual x
  #pragma unroll
  for (int mf = 0; mf < 2; ++mf) {
    #pragma unroll
    for (int nf = 0; nf < 4; ++nf) {
      const int col  = nBase + wn * 64 + nf * 16 + (lane & 15);
      const int row0 = mBase + wm * 32 + mf * 16 + ((lane >> 4) << 3);
      #pragma unroll
      for (int v = 0; v < 8; ++v) {
        const int row = row0 + v;
        out[row * OUT_DIM + col] = acc[mf][nf][v] + x[row * IN_DIM + col];
      }
    }
  }
}

extern "C" void kernel_launch(void* const* d_in, const int* in_sizes, int n_in,
                              void* d_out, int out_size, void* d_ws, size_t ws_size,
                              hipStream_t stream) {
  (void)in_sizes; (void)n_in; (void)out_size; (void)ws_size;
  const float* x      = (const float*)d_in[0];
  const float* coeffs = (const float*)d_in[1];
  const float* bw     = (const float*)d_in[2];
  float* out = (float*)d_out;

  unsigned short* Wt = (unsigned short*)d_ws;                                   // 8 MB
  float* xT = (float*)((char*)d_ws + (size_t)OUT_DIM * KTOT * sizeof(unsigned short)); // +16.8 MB

  pack_weights<<<(OUT_DIM * KTOT) / 256, 256, 0, stream>>>(coeffs, bw, Wt);
  transpose_x<<<(BATCH * IN_DIM) / 256, 256, 0, stream>>>(x, xT);
  kan_fused_gemm<<<dim3(BATCH / BM, OUT_DIM / BN), 256, 0, stream>>>(x, xT, Wt, out);
}
